// FLPVT2_14370960572502
// MI455X (gfx1250) — matrix operand order repaired and checked by reading, not compiled, not run on hardware
//
#include <hip/hip_runtime.h>
#include <hip/hip_bf16.h>

// ---------------------------------------------------------------------------
// FLatten / focused linear attention (PVT2 block) for MI455X (gfx1250).
// All dense GEMMs routed through v_wmma_f32_16x16x32_f16 (f32 accumulate),
// double-buffered LDS staging, async global->LDS copies where no conversion
// is needed (GLOBAL_LOAD_ASYNC_TO_LDS_B128, ASYNCcnt-tracked).
// Fixed problem size: B=4, N=16384 (128x128), C=512, heads=8, d=64, KS=5.
// ---------------------------------------------------------------------------

typedef __attribute__((ext_vector_type(16))) _Float16 v16h;
typedef __attribute__((ext_vector_type(8)))  _Float16 v8h;
typedef __attribute__((ext_vector_type(8)))  float    v8f;
typedef __attribute__((ext_vector_type(4)))  int      v4i;

#define BATCH   4
#define NTOK    16384
#define CDIM    512
#define HEADS   8
#define HDIM    64
#define IMGH    128
#define IMGW    128

#if __has_builtin(__builtin_amdgcn_global_load_async_to_lds_b128)
#define FL_ASYNC 1
#else
#define FL_ASYNC 0
#endif

#if FL_ASYNC
__device__ __forceinline__ void async_b128(const void* g, void* l) {
  // Per-lane 16B global -> LDS, tracked on ASYNCcnt.
  // Builtin takes generic int4* pointers (src global, dst LDS), imm offset, cpol.
  __builtin_amdgcn_global_load_async_to_lds_b128((v4i*)g, (v4i*)l, 0, 0);
}
__device__ __forceinline__ void wait_async0() {
#if __has_builtin(__builtin_amdgcn_s_wait_asynccnt)
  __builtin_amdgcn_s_wait_asynccnt(0);
#else
  asm volatile("s_wait_asynccnt 0" ::: "memory");
#endif
}
#else
__device__ __forceinline__ void wait_async0() {}
#endif

__device__ __forceinline__ v8f wmma_f16(v16h a, v16h b, v8f c) {
  // 8 args: (neg_a, A, neg_b, B, c_mod, C, reuse_a, reuse_b)
  return __builtin_amdgcn_wmma_f32_16x16x32_f16(false, a, false, b, (short)0, c,
                                                false, false);
}

// Build a 16-half fragment from two 16B LDS chunks at base and base+gap.
// A-matrix (16x32 f16): per-lane chunks K[kb..kb+8) and K[kb+16..kb+24) -> gap=16.
// B-matrix (32x16 f16): same K chunks as A                              -> gap=16.
__device__ __forceinline__ v16h ld_frag(const _Float16* base, int gap) {
  v8h lo = *(const v8h*)(base);
  v8h hi = *(const v8h*)(base + gap);
  return __builtin_shufflevector(lo, hi, 0, 1, 2, 3, 4, 5, 6, 7,
                                         8, 9, 10, 11, 12, 13, 14, 15);
}

// ---------------------------------------------------------------------------
// Kernel: f32 -> f16 convert (weights)
// ---------------------------------------------------------------------------
__global__ void flpvt2_cvt16(const float* __restrict__ src,
                             _Float16* __restrict__ dst, int n) {
  int i = blockIdx.x * 256 + threadIdx.x;
  if (i < n) dst[i] = (_Float16)src[i];
}

// ---------------------------------------------------------------------------
// Kernel: C[M,Nout] = A[M,K](f32) @ W16[Nout,K]^T + bias   (NT GEMM, WMMA)
// Block tile 128x128x32, 256 threads = 8 waves, wave tile 32x64.
// Double-buffered LDS; B tiles staged via async-to-LDS (overlaps WMMA).
// ---------------------------------------------------------------------------
__global__ void __launch_bounds__(256)
flpvt2_gemm_nt(const float* __restrict__ A, const _Float16* __restrict__ W16,
               const float* __restrict__ bias, float* __restrict__ C,
               int M, int Nout, int K) {
  __shared__ _Float16 As[2][128 * 40];   // row stride 40 halves = 80B
  __shared__ _Float16 Bs[2][128 * 40];

  const int tid = threadIdx.x;
  const int w = tid >> 5, l = tid & 31;
  const int l15 = l & 15;
  const int kbA = (l & 16) ? 8 : 0;
  const int kbB = (l & 16) ? 8 : 0;
  const int roff = (l & 16) ? 8 : 0;

  const int mwave = w & 3, nwave = w >> 2;
  const int m0 = mwave * 32, n0 = nwave * 64;

  const long arow = (long)blockIdx.y * 128;
  const long brow = (long)blockIdx.x * 128;

  v8f acc[2][4] = {};
  const int nk = K >> 5;

  // ---- prologue: stage k-step 0 into buffer 0
  for (int j = 0; j < 4; ++j) {
    int i = tid + j * 256;
    int r = i >> 3, c = (i & 7) << 2;
    float4 f = *(const float4*)(A + (arow + r) * K + c);
    _Float16* d = &As[0][r * 40 + c];
    d[0] = (_Float16)f.x; d[1] = (_Float16)f.y;
    d[2] = (_Float16)f.z; d[3] = (_Float16)f.w;
  }
  for (int j = 0; j < 2; ++j) {
    int i = tid + j * 256;
    int r = i >> 2, c = (i & 3) << 3;
#if FL_ASYNC
    async_b128(W16 + (brow + r) * K + c, &Bs[0][r * 40 + c]);
#else
    *(v8h*)&Bs[0][r * 40 + c] = *(const v8h*)(W16 + (brow + r) * K + c);
#endif
  }
  wait_async0();
  __syncthreads();

  for (int ki = 0; ki < nk; ++ki) {
    const int cur = ki & 1, nxt = cur ^ 1;
    const bool pre = (ki + 1 < nk);
    const int ktn = (ki + 1) << 5;

    // Issue next-step global traffic before touching the current tile.
    float4 fa[4];
    if (pre) {
      for (int j = 0; j < 4; ++j) {
        int i = tid + j * 256;
        int r = i >> 3, c = (i & 7) << 2;
        fa[j] = *(const float4*)(A + (arow + r) * K + ktn + c);
      }
#if FL_ASYNC
      for (int j = 0; j < 2; ++j) {
        int i = tid + j * 256;
        int r = i >> 2, c = (i & 3) << 3;
        async_b128(W16 + (brow + r) * K + ktn + c, &Bs[nxt][r * 40 + c]);
      }
#endif
    }
#if !FL_ASYNC
    v8h bregs[2];
    if (pre)
      for (int j = 0; j < 2; ++j) {
        int i = tid + j * 256;
        int r = i >> 2, c = (i & 3) << 3;
        bregs[j] = *(const v8h*)(W16 + (brow + r) * K + ktn + c);
      }
#endif

    // Compute on the current buffer.
    v16h af[2], bf[4];
    for (int mi = 0; mi < 2; ++mi)
      af[mi] = ld_frag(&As[cur][(m0 + mi * 16 + l15) * 40 + kbA], 16);
    for (int ni = 0; ni < 4; ++ni)
      bf[ni] = ld_frag(&Bs[cur][(n0 + ni * 16 + l15) * 40 + kbB], 16);
    for (int mi = 0; mi < 2; ++mi)
      for (int ni = 0; ni < 4; ++ni)
        acc[mi][ni] = wmma_f16(af[mi], bf[ni], acc[mi][ni]);

    // Drain next-step A (convert f32->f16) into the other buffer.
    if (pre) {
      for (int j = 0; j < 4; ++j) {
        int i = tid + j * 256;
        int r = i >> 3, c = (i & 7) << 2;
        _Float16* d = &As[nxt][r * 40 + c];
        d[0] = (_Float16)fa[j].x; d[1] = (_Float16)fa[j].y;
        d[2] = (_Float16)fa[j].z; d[3] = (_Float16)fa[j].w;
      }
#if !FL_ASYNC
      for (int j = 0; j < 2; ++j) {
        int i = tid + j * 256;
        int r = i >> 2, c = (i & 3) << 3;
        *(v8h*)&Bs[nxt][r * 40 + c] = bregs[j];
      }
#endif
    }
    wait_async0();
    __syncthreads();
  }

  for (int mi = 0; mi < 2; ++mi)
    for (int ni = 0; ni < 4; ++ni) {
      int col = (int)brow + n0 + ni * 16 + l15;
      float bv = bias ? bias[col] : 0.0f;
      for (int r = 0; r < 8; ++r) {
        long row = arow + m0 + mi * 16 + r + roff;
        C[row * Nout + col] = acc[mi][ni][r] + bv;
      }
    }
}

// ---------------------------------------------------------------------------
// Kernel: per-token focusing transform.
// q32[B,N,C], kv32[B,N,2C] -> q16/k16/v16 in [B,h,N,d] f16 layout.
// ---------------------------------------------------------------------------
__global__ void __launch_bounds__(256)
flpvt2_focus(const float* __restrict__ q32, const float* __restrict__ kv32,
             const float* __restrict__ ffac, const float* __restrict__ scale,
             _Float16* __restrict__ q16, _Float16* __restrict__ k16,
             _Float16* __restrict__ v16) {
  const long token = blockIdx.x;
  const int tid = threadIdx.x;
  const int b = (int)(token >> 14);
  const int n = (int)(token & 16383);

  __shared__ float4 red[256];

  const float* qrow  = q32 + token * CDIM;
  const float* kvrow = kv32 + token * (2 * CDIM);

  float qf[2], kf[2], vv[2];
  float4 part = {0.f, 0.f, 0.f, 0.f};
  for (int j = 0; j < 2; ++j) {
    int c = tid + j * 256;
    float s = log1pf(expf(scale[c]));           // softplus
    float ff = ffac[c];
    float qq = (fmaxf(qrow[c], 0.f) + 1e-6f) / s;
    float kk = (fmaxf(kvrow[c], 0.f) + 1e-6f) / s;
    vv[j] = kvrow[CDIM + c];
    float qp = exp2f(ff * log2f(qq));           // qq^ff (qq > 0)
    float kp = exp2f(ff * log2f(kk));
    qf[j] = qp; kf[j] = kp;
    part.x += qq * qq; part.y += qp * qp;
    part.z += kk * kk; part.w += kp * kp;
  }
  red[tid] = part;
  __syncthreads();
  for (int s2 = 128; s2 > 0; s2 >>= 1) {
    if (tid < s2) {
      float4 a = red[tid], bb = red[tid + s2];
      a.x += bb.x; a.y += bb.y; a.z += bb.z; a.w += bb.w;
      red[tid] = a;
    }
    __syncthreads();
  }
  float4 tot = red[0];
  float qscale = sqrtf(tot.x / tot.y);   // ||q|| / ||q^ff||
  float kscale = sqrtf(tot.z / tot.w);

  for (int j = 0; j < 2; ++j) {
    int c = tid + j * 256;
    int hh = c >> 6, dd = c & 63;
    long o = (((long)b * HEADS + hh) * NTOK + n) * HDIM + dd;
    q16[o] = (_Float16)(qf[j] * qscale);
    k16[o] = (_Float16)(kf[j] * kscale);
    v16[o] = (_Float16)vv[j];
  }
}

// ---------------------------------------------------------------------------
// Kernel: kv_mat = (k^T v)/N per (b,h) + k_mean.
// One block per bh. Double-buffered LDS transpose staging with register
// prefetch of the next 32-row chunk; one barrier per chunk.
// ---------------------------------------------------------------------------
__global__ void __launch_bounds__(256)
flpvt2_kvmat(const _Float16* __restrict__ k16, const _Float16* __restrict__ v16,
             _Float16* __restrict__ Pt16, float* __restrict__ kmean) {
  const int bh = blockIdx.x;
  const int tid = threadIdx.x;
  const int w = tid >> 5, l = tid & 31;
  const int l15 = l & 15;
  const int kbA = (l & 16) ? 8 : 0;
  const int kbB = (l & 16) ? 8 : 0;
  const int roff = (l & 16) ? 8 : 0;

  __shared__ _Float16 kT[2][64 * 40];  // [dd][n-chunk], stride 40 halves
  __shared__ _Float16 vT[2][64 * 40];  // [e ][n-chunk]

  const _Float16* kb = k16 + (long)bh * NTOK * HDIM;
  const _Float16* vb = v16 + (long)bh * NTOK * HDIM;

  int mi[2], ni[2];
  for (int j = 0; j < 2; ++j) {
    int t = w * 2 + j;
    mi[j] = t >> 2;
    ni[j] = t & 3;
  }
  v8f acc[2] = {};
  float ksum = 0.f;

  const int srcR = tid >> 3;         // chunk row 0..31
  const int srcC = (tid & 7) << 3;   // dd group base
  const int nchunk = NTOK / 32;

  // prologue: chunk 0 -> buffer 0
  {
    v8h kc = *(const v8h*)(kb + (long)srcR * HDIM + srcC);
    v8h vc = *(const v8h*)(vb + (long)srcR * HDIM + srcC);
    for (int j2 = 0; j2 < 8; ++j2) {
      kT[0][(srcC + j2) * 40 + srcR] = kc[j2];
      vT[0][(srcC + j2) * 40 + srcR] = vc[j2];
    }
  }
  __syncthreads();

  for (int ci = 0; ci < nchunk; ++ci) {
    const int cur = ci & 1, nxt = cur ^ 1;
    const bool pre = (ci + 1 < nchunk);
    v8h kn, vn;
    if (pre) {
      long base = (long)((ci + 1) * 32 + srcR) * HDIM + srcC;
      kn = *(const v8h*)(kb + base);
      vn = *(const v8h*)(vb + base);
    }

    if (tid < 64) {                       // fused k-mean partial
      float s = 0.f;
      for (int j2 = 0; j2 < 32; ++j2) s += (float)kT[cur][tid * 40 + j2];
      ksum += s;
    }
    for (int j = 0; j < 2; ++j) {
      v16h af = ld_frag(&kT[cur][(mi[j] * 16 + l15) * 40 + kbA], 16);
      v16h bf = ld_frag(&vT[cur][(ni[j] * 16 + l15) * 40 + kbB], 16);
      acc[j] = wmma_f16(af, bf, acc[j]);
    }

    if (pre)
      for (int j2 = 0; j2 < 8; ++j2) {    // transpose-scatter next chunk
        kT[nxt][(srcC + j2) * 40 + srcR] = kn[j2];
        vT[nxt][(srcC + j2) * 40 + srcR] = vn[j2];
      }
    __syncthreads();
  }

  const float invN = 1.0f / (float)NTOK;
  for (int j = 0; j < 2; ++j)
    for (int r = 0; r < 8; ++r) {
      int dd = mi[j] * 16 + r + roff;
      int e  = ni[j] * 16 + l15;
      Pt16[(long)bh * (HDIM * HDIM) + e * HDIM + dd] =
          (_Float16)(acc[j][r] * invN);
    }
  if (tid < 64) kmean[bh * HDIM + tid] = ksum * invN;
}

// ---------------------------------------------------------------------------
// Kernel: x_attn[b,n,h*64+e] = z_n * (q[n,:] @ P[:,e])  per (b,h), 128-row tiles.
// q/P tiles staged to LDS via async-to-LDS when available.
// ---------------------------------------------------------------------------
__global__ void __launch_bounds__(256)
flpvt2_attn(const _Float16* __restrict__ q16, const _Float16* __restrict__ Pt16,
            const float* __restrict__ kmean, float* __restrict__ y) {
  const int nb = blockIdx.x, bh = blockIdx.y;
  const int b = bh >> 3, hh = bh & 7;
  const int tid = threadIdx.x;
  const int w = tid >> 5, l = tid & 31;
  const int l15 = l & 15;
  const int kbA = (l & 16) ? 8 : 0;
  const int kbB = (l & 16) ? 8 : 0;
  const int roff = (l & 16) ? 8 : 0;

  __shared__ _Float16 qls[128 * 72];  // [n-local][dd], stride 72 halves (144B)
  __shared__ _Float16 Pls[64 * 72];   // [e][dd]
  __shared__ float kml[64];
  __shared__ float zls[128];

  const _Float16* qb = q16 + ((long)bh * NTOK + (long)nb * 128) * HDIM;
  const _Float16* pb = Pt16 + (long)bh * (HDIM * HDIM);
  for (int j = 0; j < 4; ++j) {       // 128x64 halves
    int i = tid + j * 256;
    int r = i >> 3, c = (i & 7) << 3;
#if FL_ASYNC
    async_b128(qb + (long)r * HDIM + c, &qls[r * 72 + c]);
#else
    *(v8h*)&qls[r * 72 + c] = *(const v8h*)(qb + (long)r * HDIM + c);
#endif
  }
  for (int j = 0; j < 2; ++j) {       // 64x64 halves
    int i = tid + j * 256;
    int r = i >> 3, c = (i & 7) << 3;
#if FL_ASYNC
    async_b128(pb + (long)r * HDIM + c, &Pls[r * 72 + c]);
#else
    *(v8h*)&Pls[r * 72 + c] = *(const v8h*)(pb + (long)r * HDIM + c);
#endif
  }
  if (tid < 64) kml[tid] = kmean[bh * HDIM + tid];
  wait_async0();
  __syncthreads();

  if (tid < 128) {                    // z = 1/(q . k_mean + eps)
    float dot = 0.f;
    for (int dd = 0; dd < HDIM; ++dd)
      dot += (float)qls[tid * 72 + dd] * kml[dd];
    zls[tid] = 1.0f / (dot + 1e-6f);
  }
  __syncthreads();

  const int m0 = w * 16;              // wave owns 16 rows
  v16h af[2];
  for (int ks = 0; ks < 2; ++ks)
    af[ks] = ld_frag(&qls[(m0 + l15) * 72 + ks * 32 + kbA], 16);

  v8f acc[4] = {};
  for (int ni = 0; ni < 4; ++ni)
    for (int ks = 0; ks < 2; ++ks) {
      v16h bf = ld_frag(&Pls[(ni * 16 + l15) * 72 + ks * 32 + kbB], 16);
      acc[ni] = wmma_f16(af[ks], bf, acc[ni]);
    }

  float* yb = y + ((long)b * NTOK + (long)nb * 128) * CDIM + hh * HDIM;
  for (int ni = 0; ni < 4; ++ni) {
    int e = ni * 16 + l15;
    for (int r = 0; r < 8; ++r) {
      int m = m0 + r + roff;
      yb[(long)m * CDIM + e] = acc[ni][r] * zls[m];
    }
  }
}

// ---------------------------------------------------------------------------
// Kernel: 5x5 depthwise conv on v (read from kv32 layout), accumulate into y.
// block (64 ch, 4 rows); tile 4 rows x 16 cols; LDS halo 8x20x64 f32 = 40 KB.
// ---------------------------------------------------------------------------
__global__ void __launch_bounds__(256)
flpvt2_dwconv(const float* __restrict__ kv32, const float* __restrict__ dwc_w,
              const float* __restrict__ dwc_b, float* __restrict__ y) {
  const int dd = threadIdx.x;        // channel 0..63
  const int ty = threadIdx.y;        // row in tile 0..3
  const int tid = ty * 64 + dd;
  const int x0 = blockIdx.x * 16, y0 = blockIdx.y * 4;
  const int bh = blockIdx.z;
  const int b = bh >> 3, hh = bh & 7;

  __shared__ float vt[8][20][64];

  const int quad = tid >> 6, ch = tid & 63;
  const float* src = kv32 + (long)b * NTOK * (2 * CDIM) + CDIM + hh * HDIM + ch;
  for (int it = 0; it < 40; ++it) {  // 160 (row,col) halo cells x 64 ch
    int p = it * 4 + quad;
    int rr = p / 20, cc = p % 20;
    int gy = y0 + rr - 2, gx = x0 + cc - 2;
    float val = 0.f;
    if (gy >= 0 && gy < IMGH && gx >= 0 && gx < IMGW)
      val = src[(long)(gy * IMGW + gx) * (2 * CDIM)];
    vt[rr][cc][ch] = val;
  }
  __syncthreads();

  float wgt[25];
  for (int i = 0; i < 25; ++i) wgt[i] = dwc_w[dd * 25 + i];
  const float bias = dwc_b[dd];

  float* yb = y + (long)b * NTOK * CDIM + hh * HDIM + dd;
  for (int xx = 0; xx < 16; ++xx) {
    float s = bias;
    for (int ky = 0; ky < 5; ++ky)
      for (int kx = 0; kx < 5; ++kx)
        s += vt[ty + ky][xx + kx][dd] * wgt[ky * 5 + kx];
    long n = (long)(y0 + ty) * IMGW + (x0 + xx);
    yb[n * CDIM] += s;               // unique owner -> plain RMW is fine
  }
}

// ---------------------------------------------------------------------------
// Host-side launch
// ---------------------------------------------------------------------------
extern "C" void kernel_launch(void* const* d_in, const int* in_sizes, int n_in,
                              void* d_out, int out_size, void* d_ws, size_t ws_size,
                              hipStream_t stream) {
  const float* x     = (const float*)d_in[0];
  const float* Wq    = (const float*)d_in[1];
  const float* bq    = (const float*)d_in[2];
  const float* Wkv   = (const float*)d_in[3];
  const float* bkv   = (const float*)d_in[4];
  const float* Wp    = (const float*)d_in[5];
  const float* bp    = (const float*)d_in[6];
  const float* ffac  = (const float*)d_in[7];
  const float* scale = (const float*)d_in[8];
  const float* dwc_w = (const float*)d_in[9];
  const float* dwc_b = (const float*)d_in[10];

  const long M = (long)BATCH * NTOK;       // 65536 token rows

  size_t off = 0;
  auto alloc = [&](size_t bytes) {
    void* p = (char*)d_ws + off;
    off += (bytes + 255) & ~(size_t)255;
    return p;
  };
  float*     q32   = (float*)alloc(M * CDIM * 4);          // 134 MB
  float*     kv32  = (float*)alloc(M * 2 * CDIM * 4);      // 268 MB
  float*     yattn = (float*)alloc(M * CDIM * 4);          // 134 MB
  _Float16*  q16   = (_Float16*)alloc(M * CDIM * 2);       //  67 MB
  _Float16*  k16   = (_Float16*)alloc(M * CDIM * 2);
  _Float16*  v16   = (_Float16*)alloc(M * CDIM * 2);
  _Float16*  Pt16  = (_Float16*)alloc((size_t)32 * HDIM * HDIM * 2);
  float*     kmean = (float*)alloc((size_t)32 * HDIM * 4);
  _Float16*  Wq16  = (_Float16*)alloc((size_t)CDIM * CDIM * 2);
  _Float16*  Wkv16 = (_Float16*)alloc((size_t)2 * CDIM * CDIM * 2);
  _Float16*  Wp16  = (_Float16*)alloc((size_t)CDIM * CDIM * 2);

  // 1) weights -> f16
  flpvt2_cvt16<<<(CDIM * CDIM + 255) / 256, 256, 0, stream>>>(Wq, Wq16, CDIM * CDIM);
  flpvt2_cvt16<<<(2 * CDIM * CDIM + 255) / 256, 256, 0, stream>>>(Wkv, Wkv16, 2 * CDIM * CDIM);
  flpvt2_cvt16<<<(CDIM * CDIM + 255) / 256, 256, 0, stream>>>(Wp, Wp16, CDIM * CDIM);

  // 2) q = x @ Wq^T + bq ; kv = x @ Wkv^T + bkv
  flpvt2_gemm_nt<<<dim3(CDIM / 128, M / 128), 256, 0, stream>>>(
      x, Wq16, bq, q32, (int)M, CDIM, CDIM);
  flpvt2_gemm_nt<<<dim3(2 * CDIM / 128, M / 128), 256, 0, stream>>>(
      x, Wkv16, bkv, kv32, (int)M, 2 * CDIM, CDIM);

  // 3) focusing transform -> q16/k16/v16 (head-major)
  flpvt2_focus<<<(int)M, 256, 0, stream>>>(q32, kv32, ffac, scale, q16, k16, v16);

  // 4) kv_mat + k_mean per (b,h)
  flpvt2_kvmat<<<BATCH * HEADS, 256, 0, stream>>>(k16, v16, Pt16, kmean);

  // 5) attention apply -> yattn
  flpvt2_attn<<<dim3(NTOK / 128, BATCH * HEADS), 256, 0, stream>>>(
      q16, Pt16, kmean, yattn);

  // 6) depthwise conv accumulate into yattn
  flpvt2_dwconv<<<dim3(IMGW / 16, IMGH / 4, BATCH * HEADS), dim3(64, 4), 0, stream>>>(
      kv32, dwc_w, dwc_b, yattn);

  // 7) out = yattn @ Wp^T + bp
  flpvt2_gemm_nt<<<dim3(CDIM / 128, M / 128), 256, 0, stream>>>(
      yattn, Wp16, bp, (float*)d_out, (int)M, CDIM, CDIM);
}
